// HeteroRGCNLayer_15994458210645
// MI455X (gfx1250) — matrix operand order repaired and checked by reading, not compiled, not run on hardware
//
#include <hip/hip_runtime.h>

#define NNODES 50000
#define NEDGES 500000
#define D 256   // D_IN == D_OUT

typedef __attribute__((ext_vector_type(16))) __bf16 v16bf;
typedef __attribute__((ext_vector_type(8)))  float  v8f;

// ---------------------------------------------------------------------------
// utility: zero a float buffer (grid-stride)
// ---------------------------------------------------------------------------
__global__ void zero_kernel(float* __restrict__ p, int n) {
    int i = blockIdx.x * blockDim.x + threadIdx.x;
    int stride = gridDim.x * blockDim.x;
    for (; i < n; i += stride) p[i] = 0.0f;
}

// ---------------------------------------------------------------------------
// transpose W (k-major [D_IN, D_OUT]) into column-major and split f32 into
// bf16 hi + bf16 lo residual:  Wt*[n*D + k] correspond to W[k*D + n]
// ---------------------------------------------------------------------------
__global__ void prep_w_kernel(const float* __restrict__ W,
                              __bf16* __restrict__ WtHi,
                              __bf16* __restrict__ WtLo) {
    int i = blockIdx.x * blockDim.x + threadIdx.x;
    if (i >= D * D) return;
    int n = i / D, k = i % D;
    float w  = W[k * D + n];
    __bf16 h = (__bf16)w;
    WtHi[i] = h;
    WtLo[i] = (__bf16)(w - (float)h);
}

// ---------------------------------------------------------------------------
// edge scatter: Acc[dst] += X[src]  (256 f32 per edge, float4 granularity),
// cnt[dst] += 1.  Hardware f32 atomics (global_atomic_add_f32).
// One thread per (edge, 4 contiguous floats): idx = e*64 + q.
// ---------------------------------------------------------------------------
__global__ void scatter_kernel(const float* __restrict__ feat,
                               const int* __restrict__ src,
                               const int* __restrict__ dst,
                               float* __restrict__ Acc,
                               float* __restrict__ cnt) {
    int i = blockIdx.x * blockDim.x + threadIdx.x;   // < NEDGES*64 = 32e6
    if (i >= NEDGES * 64) return;
    int e = i >> 6;
    int q = i & 63;
    int s = src[e];
    int d = dst[e];
    float4 v = ((const float4*)feat)[s * 64 + q];
    float* p = Acc + (long long)d * D + q * 4;
    unsafeAtomicAdd(p + 0, v.x);
    unsafeAtomicAdd(p + 1, v.y);
    unsafeAtomicAdd(p + 2, v.z);
    unsafeAtomicAdd(p + 3, v.w);
    if (q == 0) unsafeAtomicAdd(cnt + d, 1.0f);
}

// ---------------------------------------------------------------------------
// node-space GEMM with bf16x3 split precision:
//   out[m, n] += (Acc[m,:] / max(cnt[m],1)) @ W[:, n] + b[n] * 1[cnt[m] > 0]
// One wave computes a 16(M) x 64(N) strip = 4 WMMA tiles, reusing A frags.
// Fragment layouts used here (contiguous k, same order for A and B):
//   A 16x32 bf16 : lanes 0-15 row M=lane, elems = K{kb..kb+15};
//                  lanes 16-31 same rows,  elems = K{kb+16..kb+31}
//   B 32x16 bf16 : lanes 0-15 col N=lane, elems = K{kb..kb+15};
//                  lanes 16-31 same cols, elems = K{kb+16..kb+31}
//   C 16x16 f32  : elem i -> row m0 + 8*(lane>>4) + i, col n0 + (lane&15)
// ---------------------------------------------------------------------------
__global__ void gemm_kernel(const float* __restrict__ Acc,
                            const float* __restrict__ cnt,
                            const __bf16* __restrict__ WtHi,
                            const __bf16* __restrict__ WtLo,
                            const float* __restrict__ bias,
                            float* __restrict__ out) {
    const int WAVES = (NNODES / 16) * (D / 64);       // 3125 * 4 = 12500
    int wave = (blockIdx.x * blockDim.x + threadIdx.x) >> 5;
    int lane = threadIdx.x & 31;
    if (wave >= WAVES) return;                         // whole waves exit

    int mt = wave >> 2;
    int ng = wave & 3;
    int m0 = mt * 16;
    int n0 = ng * 64;

    int half = lane >> 4;      // which half-wave
    int lr   = lane & 15;

    int   arow = m0 + lr;                              // A row for this lane
    float inv  = 1.0f / fmaxf(cnt[arow], 1.0f);

    v8f acc[4] = {};

    for (int kb = 0; kb < D; kb += 32) {
        // ---- build split-precision A fragment from f32 Acc ----
        const float* ab = Acc + (long long)arow * D + kb + half * 16;
        float4 a0 = *(const float4*)(ab + 0);
        float4 a1 = *(const float4*)(ab + 4);
        float4 a2 = *(const float4*)(ab + 8);
        float4 a3 = *(const float4*)(ab + 12);
        float af[16] = {a0.x, a0.y, a0.z, a0.w, a1.x, a1.y, a1.z, a1.w,
                        a2.x, a2.y, a2.z, a2.w, a3.x, a3.y, a3.z, a3.w};
        v16bf ah, al;
        #pragma unroll
        for (int t = 0; t < 16; ++t) {
            float x  = af[t] * inv;
            __bf16 h = (__bf16)x;
            ah[t] = h;
            al[t] = (__bf16)(x - (float)h);
        }

        // ---- 4 N-tiles reuse the A fragment ----
        #pragma unroll
        for (int nt = 0; nt < 4; ++nt) {
            int col = n0 + nt * 16 + lr;
            long long boff = (long long)col * D + kb + half * 16;
            v16bf bh = *(const v16bf*)(WtHi + boff);
            v16bf bl = *(const v16bf*)(WtLo + boff);
            // bf16x3: Ah*Bh + Al*Bh + Ah*Bl, f32 accumulate
            acc[nt] = __builtin_amdgcn_wmma_f32_16x16x32_bf16(
                false, ah, false, bh, (short)0, acc[nt], false, false);
            acc[nt] = __builtin_amdgcn_wmma_f32_16x16x32_bf16(
                false, al, false, bh, (short)0, acc[nt], false, false);
            acc[nt] = __builtin_amdgcn_wmma_f32_16x16x32_bf16(
                false, ah, false, bl, (short)0, acc[nt], false, false);
        }
    }

    // ---- epilogue: add bias * 1[cnt>0], accumulate into out ----
    float ind[8];
    #pragma unroll
    for (int i = 0; i < 8; ++i) {
        int row = m0 + half * 8 + i;
        ind[i] = (cnt[row] >= 1.0f) ? 1.0f : 0.0f;
    }
    #pragma unroll
    for (int nt = 0; nt < 4; ++nt) {
        int col = n0 + nt * 16 + lr;
        float bv = bias[col];
        #pragma unroll
        for (int i = 0; i < 8; ++i) {
            int row = m0 + half * 8 + i;
            float* p = out + (long long)row * D + col;
            *p += acc[nt][i] + bv * ind[i];
        }
    }
}

// ---------------------------------------------------------------------------
__global__ void relu_kernel(float* __restrict__ p, int n) {
    int i = blockIdx.x * blockDim.x + threadIdx.x;
    int stride = gridDim.x * blockDim.x;
    for (; i < n; i += stride) p[i] = fmaxf(p[i], 0.0f);
}

// ---------------------------------------------------------------------------
extern "C" void kernel_launch(void* const* d_in, const int* in_sizes, int n_in,
                              void* d_out, int out_size, void* d_ws, size_t ws_size,
                              hipStream_t stream) {
    const float* feat   = (const float*)d_in[0];
    const int*   src[3] = {(const int*)d_in[1], (const int*)d_in[3], (const int*)d_in[5]};
    const int*   dst[3] = {(const int*)d_in[2], (const int*)d_in[4], (const int*)d_in[6]};
    const float* W[3]   = {(const float*)d_in[7], (const float*)d_in[9],  (const float*)d_in[11]};
    const float* b[3]   = {(const float*)d_in[8], (const float*)d_in[10], (const float*)d_in[12]};
    float* out = (float*)d_out;

    // workspace layout: Acc[NNODES*D] f32 | cnt[NNODES] f32 | WtHi | WtLo (bf16)
    char*  ws       = (char*)d_ws;
    size_t accElems = (size_t)NNODES * D;              // 12.8M floats
    float* Acc = (float*)ws;
    float* cnt = Acc + accElems;                       // contiguous after Acc
    size_t off = (((accElems + NNODES) * 4) + 255) & ~(size_t)255;
    __bf16* WtHi = (__bf16*)(ws + off);
    __bf16* WtLo = WtHi + D * D;

    const int outN = NNODES * D;
    zero_kernel<<<4096, 256, 0, stream>>>(out, outN);

    for (int r = 0; r < 3; ++r) {
        // zero Acc + cnt in one pass (contiguous)
        zero_kernel<<<4096, 256, 0, stream>>>(Acc, (int)(accElems + NNODES));
        prep_w_kernel<<<(D * D + 255) / 256, 256, 0, stream>>>(W[r], WtHi, WtLo);

        int total = NEDGES * 64;                       // 32,000,000
        scatter_kernel<<<(total + 255) / 256, 256, 0, stream>>>(
            feat, src[r], dst[r], Acc, cnt);

        int waves = (NNODES / 16) * (D / 64);          // 12500
        gemm_kernel<<<(waves + 7) / 8, 256, 0, stream>>>(
            Acc, cnt, WtHi, WtLo, b[r], out);
    }

    relu_kernel<<<4096, 256, 0, stream>>>(out, outN);
}
